// MyRNN_69526930588151
// MI455X (gfx1250) — matrix operand order repaired and checked by reading, not compiled, not run on hardware
//
#include <hip/hip_runtime.h>
#include <hip/hip_bf16.h>

typedef __attribute__((ext_vector_type(16))) _Float16 v16h;
typedef __attribute__((ext_vector_type(8)))  float    v8f;

// Problem constants (fixed by the harness setup_inputs()).
constexpr int N   = 2048;   // state size
constexpr int LT  = 4096;   // time steps
constexpr int G   = 32;     // persistent workgroups
constexpr int BLK = 128;    // 4 waves per workgroup; 4*16 = 64 rows/WG; 32*64 = 2048 rows
constexpr float DT_TAU = 0.1f;

// ---------------------------------------------------------------------------
// Prep kernel 1: zero the double-buffered x state, zero the barrier counter,
// write out[:,0] = 0 (harness poisons d_out/d_ws, and graph replays do not
// re-initialize, so this must run every launch).
// ---------------------------------------------------------------------------
__global__ void rnn_init_kernel(float* __restrict__ xbuf, int* __restrict__ counter,
                                float* __restrict__ out) {
    int i = blockIdx.x * blockDim.x + threadIdx.x;
    if (i < 2 * N) xbuf[i] = 0.0f;
    if (i < N)     out[(size_t)i * LT] = 0.0f;
    if (i == 0)
        __hip_atomic_store(counter, 0, __ATOMIC_RELAXED, __HIP_MEMORY_SCOPE_AGENT);
}

// ---------------------------------------------------------------------------
// Prep kernel 2: J = Jij * M, converted to f16 and swizzled into the exact
// WMMA 16x32 f16 A-fragment lane layout so the main loop's loads are 32
// contiguous bytes per lane (fully coalesced, 1024B per wave per tile).
//
// A-fragment k order used here (contiguous, same order as the B reads from sh_s):
//   lanes 0-15  (M=row&15): VGPR0-3 = K 0..7,   VGPR4-7 = K 8..15
//   lanes 16-31 (M=row&15): VGPR0-3 = K 16..23, VGPR4-7 = K 24..31
// Tile (m_tile, k_tile) -> 512 halfs; within tile lane L owns halfs
// [L*16, L*16+16): chunk0 (8 halfs) then chunk1 (8 halfs).
// ---------------------------------------------------------------------------
__global__ void rnn_pack_kernel(const float* __restrict__ Jij,
                                const float* __restrict__ M,
                                _Float16* __restrict__ Jh) {
    int idx = blockIdx.x * blockDim.x + threadIdx.x;   // 0 .. N*N-1
    int row = idx >> 11;          // /2048
    int k   = idx & (N - 1);
    int m_tile = row >> 4;
    int k_tile = k >> 5;
    int m      = row & 15;
    int k_in   = k & 31;
    int group  = (k_in >> 4) & 1;            // which half-wave holds this K
    int chunk  = (k_in >> 3) & 1;            // first or second 16B chunk
    int pos    = k_in & 7;
    int lane   = m + group * 16;
    size_t dst = ((size_t)m_tile * (N / 32) + k_tile) * 512 + lane * 16 + chunk * 8 + pos;
    Jh[dst] = (_Float16)(Jij[idx] * M[idx]);
}

// ---------------------------------------------------------------------------
// Main persistent kernel: 32 WGs x 128 threads, all co-resident.
// Each wave owns a 16-row tile of the output; per step it runs 64 WMMAs
// (K = 2048 in chunks of 32) with s broadcast into all 16 B columns, then
// lanes 0/16 extract column 0 of the f32 accumulator and do the Euler update.
// Grid-wide sync: monotonic atomic counter (target = G*(t+1)).
// ---------------------------------------------------------------------------
__global__ void __launch_bounds__(BLK)
rnn_scan_kernel(const float* __restrict__ II,     // [N, LT] row-major
                const _Float16* __restrict__ Jh,  // packed f16 A-fragments
                float* __restrict__ xbuf,         // [2, N] double buffer
                int* __restrict__ counter,        // grid barrier
                float* __restrict__ out) {        // [N, LT] row-major
    __shared__ __align__(64) _Float16 sh_s[N];    // sigmoid(x) staged as f16

    const int tid  = threadIdx.x;
    const int lane = tid & 31;
    const int wave = tid >> 5;
    const int row_base = blockIdx.x * 64 + wave * 16;      // 16 rows per wave
    const _Float16* wave_tiles =
        Jh + (size_t)(row_base >> 4) * (N / 32) * 512;     // this wave's tile row

    for (int t = 0; t < LT - 1; ++t) {
        const float* xc = xbuf + (t & 1) * N;
        float*       xn = xbuf + ((t + 1) & 1) * N;

        // s = sigmoid(x_t) -> LDS (f16), coalesced, whole vector per WG.
        for (int i = tid; i < N; i += BLK) {
            float xv = xc[i];
            sh_s[i] = (_Float16)(1.0f / (1.0f + __expf(-xv)));
        }
        __syncthreads();

        // y[row_base .. row_base+15] = J @ s  via WMMA, f32 accumulate.
        v8f acc0 = {};
        v8f acc1 = {};
        const int bofs = (lane & 16);   // B half-wave K offset (0 or 16 halfs)
#pragma unroll 4
        for (int kt = 0; kt < N / 32; kt += 2) {
            v16h a0 = *(const v16h*)(wave_tiles + (size_t)kt * 512 + lane * 16);
            v16h b0 = *(const v16h*)(&sh_s[kt * 32 + bofs]);
            acc0 = __builtin_amdgcn_wmma_f32_16x16x32_f16(
                false, a0, false, b0, (short)0, acc0, false, false);
            v16h a1 = *(const v16h*)(wave_tiles + (size_t)(kt + 1) * 512 + lane * 16);
            v16h b1 = *(const v16h*)(&sh_s[(kt + 1) * 32 + bofs]);
            acc1 = __builtin_amdgcn_wmma_f32_16x16x32_f16(
                false, a1, false, b1, (short)0, acc1, false, false);
        }
        v8f acc = acc0 + acc1;

        // C layout: VGPR r -> row r (lanes 0-15) / row r+8 (lanes 16-31),
        // column = lane&15; all columns identical (B was broadcast), take col 0.
        if ((lane & 15) == 0) {
            int rbase = row_base + ((lane & 16) ? 8 : 0);
#pragma unroll
            for (int r = 0; r < 8; ++r) {
                int row  = rbase + r;
                float y  = acc[r];
                float xo = xc[row];
                float xv = xo + DT_TAU * (-xo + y + II[(size_t)row * LT + t]);
                xn[row] = xv;
                out[(size_t)row * LT + t + 1] = xv;
            }
        }

        // ---- grid-wide barrier (monotonic counter, sense-free) ----
        __threadfence();       // make xn/out stores visible at device scope
        __syncthreads();       // all waves in WG arrived
        if (tid == 0) {
            __hip_atomic_fetch_add(counter, 1, __ATOMIC_ACQ_REL,
                                   __HIP_MEMORY_SCOPE_AGENT);
            const int target = G * (t + 1);
            while (__hip_atomic_load(counter, __ATOMIC_ACQUIRE,
                                     __HIP_MEMORY_SCOPE_AGENT) < target) {
                __builtin_amdgcn_s_sleep(2);
            }
        }
        __syncthreads();
        __threadfence();       // invalidate caches so next step sees fresh xbuf
    }
}

// ---------------------------------------------------------------------------
extern "C" void kernel_launch(void* const* d_in, const int* in_sizes, int n_in,
                              void* d_out, int out_size, void* d_ws, size_t ws_size,
                              hipStream_t stream) {
    const float* II  = (const float*)d_in[0];   // [2048, 4096] f32
    const float* Jij = (const float*)d_in[1];   // [2048, 2048] f32
    const float* M   = (const float*)d_in[2];   // [2048, 2048] f32
    float* out = (float*)d_out;                 // [2048, 4096] f32

    // Workspace layout: Jh (8 MB f16) | xbuf (2*N f32) | counter (int)
    _Float16* Jh  = (_Float16*)d_ws;
    float* xbuf   = (float*)((char*)d_ws + (size_t)N * N * sizeof(_Float16));
    int*   counter = (int*)((char*)d_ws + (size_t)N * N * sizeof(_Float16)
                                        + (size_t)2 * N * sizeof(float));

    rnn_init_kernel<<<(2 * N + 255) / 256, 256, 0, stream>>>(xbuf, counter, out);
    rnn_pack_kernel<<<(N * N) / 256, 256, 0, stream>>>(Jij, M, Jh);
    rnn_scan_kernel<<<G, BLK, 0, stream>>>(II, Jh, xbuf, counter, out);
}
